// PalaceAttention_42245298323779
// MI455X (gfx1250) — matrix operand order repaired and checked by reading, not compiled, not run on hardware
//
#include <hip/hip_runtime.h>
#include <hip/hip_bf16.h>

typedef __attribute__((ext_vector_type(16))) __bf16 v16bf;
typedef __attribute__((ext_vector_type(8)))  float  v8f;

#define T_SEQ   2048
#define DMODEL  1024
#define NH      16
#define HD      64
#define KT      64            // keys per iteration (4 score tiles)
#define QW      16            // q rows per wave
#define WAVES   4
#define QB      (QW * WAVES)  // 64 q rows per block

// LDS strides in bf16 elements: 72 -> 144B rows, every vector access 16B aligned
#define KV_STRIDE 72          // 64 key rows x 64 dims (+pad)
#define TR_STRIDE 72          // 64 dim rows  x 64 keys (+pad)
#define P_STRIDE  72          // 16 q rows    x 64 keys (+pad)

#define LOG2E 1.4426950408889634f

union V16U { v16bf v; __bf16 e[16]; uint4 u[2]; };
union B8U  { __bf16 e[8]; uint4 u; };

#define WMMA_BF16(A, B, C) \
  __builtin_amdgcn_wmma_f32_16x16x32_bf16(false, (A), false, (B), (short)0, (C), false, false)

// convert 8 consecutive f32 (32B-aligned) to 8 bf16
__device__ __forceinline__ void cvt8(const float* __restrict__ p, __bf16* d) {
  float4 a = *(const float4*)p;
  float4 b = *(const float4*)(p + 4);
  d[0] = (__bf16)a.x; d[1] = (__bf16)a.y; d[2] = (__bf16)a.z; d[3] = (__bf16)a.w;
  d[4] = (__bf16)b.x; d[5] = (__bf16)b.y; d[6] = (__bf16)b.z; d[7] = (__bf16)b.w;
}

// bf16 p[0..7] and p[16..23] from LDS (16B-aligned) -> WMMA B operand
__device__ __forceinline__ v16bf ldsB(const __bf16* p) {
  V16U t;
  t.u[0] = *(const uint4*)p;
  t.u[1] = *(const uint4*)(p + 16);
  return t.v;
}

// lane-xor shuffle within 16-lane rows via v_permlane16_b32 (VALU, not DS)
__device__ __forceinline__ float plane16(float v, unsigned s0, unsigned s1) {
  unsigned u = __float_as_uint(v);
  unsigned r = __builtin_amdgcn_permlane16(u, u, s0, s1, false, false);
  return __uint_as_float(r);
}
// max butterfly across a 16-lane half (C-layout rows span a half)
__device__ __forceinline__ float hmax16(float v) {
  v = fmaxf(v, plane16(v, 0x67452301u, 0xEFCDAB89u));  // xor 1
  v = fmaxf(v, plane16(v, 0x54761032u, 0xDCFE98BAu));  // xor 2
  v = fmaxf(v, plane16(v, 0x32107654u, 0xBA98FEDCu));  // xor 4
  v = fmaxf(v, plane16(v, 0xFEDCBA98u, 0x76543210u));  // xor 8
  return v;
}
__device__ __forceinline__ float fexp2(float x) { return __builtin_amdgcn_exp2f(x); }

__global__ __launch_bounds__(128)
void palace_attn_wmma(const float* __restrict__ x,
                      const float* __restrict__ wptr,
                      float* __restrict__ out) {
  __shared__ __bf16 kv_row[KT * KV_STRIDE];        // keys row-major (B of Q*K^T)
  __shared__ __bf16 kv_tr [HD * TR_STRIDE];        // keys transposed (B of P*V)
  __shared__ __bf16 p_lds [WAVES * QW * P_STRIDE]; // per-wave P (C->A relayout)

  const int tid  = threadIdx.x;
  const int lane = tid & 31;
  const int wid  = tid >> 5;
  const int l15  = lane & 15;
  const int hi   = (lane >> 4) & 1;

  const int qblocks = T_SEQ / QB;           // 32
  const int bh = blockIdx.x / qblocks;
  const int qb = blockIdx.x % qblocks;
  const int b  = bh / NH;
  const int h  = bh % NH;
  const int q0 = qb * QB + wid * QW;        // this wave's first query row

  const float scale = 0.125f;               // hd^-0.5, hd = 64
  const float wsig  = 1.0f / (1.0f + __expf(-wptr[0]));

  // palace mask * scale * log2(e): q%8 == r (VGPR row), k%8 == lane&7 (invariant)
  float mf[8];
#pragma unroll
  for (int r = 0; r < 8; ++r)
    mf[r] = ((r == (lane & 7)) ? scale : scale * wsig) * LOG2E;

  // all-ones B operand: row-sum via matrix pipe (P * ones -> every lane gets sum)
  V16U onesb;
#pragma unroll
  for (int i = 0; i < 16; ++i) onesb.e[i] = (__bf16)1.0f;

  // ---- Q tile (16x64) loaded from global directly in WMMA A layout ----
  const float* qrow = x + ((size_t)(b * T_SEQ + q0 + l15) * DMODEL + h * HD);
  const int aoff = hi ? 8 : 0;              // A layout: lanes>=16 hold K=8..15/24..31
  V16U qa[2];
#pragma unroll
  for (int s = 0; s < 2; ++s) {             // k-dim slices 0..31, 32..63
    cvt8(qrow + s * 32 + aoff,      qa[s].e);
    cvt8(qrow + s * 32 + aoff + 16, qa[s].e + 8);
  }

  // flash-attention running state (registers); softmax kept in base-2 domain
  v8f o0 = {}, o1 = {}, o2 = {}, o3 = {};
  v8f ls = {};                              // unnormalized row-sum accumulator
  float mrow[8];
#pragma unroll
  for (int r = 0; r < 8; ++r) mrow[r] = -1e30f;

  const int doff = hi ? 8 : 0;              // B layout: lanes>=16 hold K=8..15/24..31

#pragma unroll 1
  for (int k0 = 0; k0 < T_SEQ; k0 += KT) {
    __syncthreads();                        // previous tile fully consumed
    { // cooperative K/V load: 64 keys x 64 dims; thread = 4 keys x 8 dims
      const int kr0 = (tid & 15) * 4;       // key group
      const int d0  = (tid >> 4) * 8;       // dim segment
      const float* src = x + ((size_t)(b * T_SEQ + k0 + kr0) * DMODEL
                              + h * HD + d0);
      if (k0 + KT < T_SEQ)
        __builtin_prefetch(src + (size_t)KT * DMODEL, 0, 3); // global_prefetch_b8
      B8U t[4];
#pragma unroll
      for (int kk = 0; kk < 4; ++kk) {
        cvt8(src + (size_t)kk * DMODEL, t[kk].e);
        *(uint4*)&kv_row[(kr0 + kk) * KV_STRIDE + d0] = t[kk].u;   // b128 store
      }
#pragma unroll
      for (int d = 0; d < 8; ++d) {         // transposed: pack 4 keys per b64
        union { __bf16 e[4]; uint2 u; } q4;
        q4.e[0] = t[0].e[d]; q4.e[1] = t[1].e[d];
        q4.e[2] = t[2].e[d]; q4.e[3] = t[3].e[d];
        *(uint2*)&kv_tr[(d0 + d) * TR_STRIDE + kr0] = q4.u;        // b64 store
      }
    }
    __syncthreads();

    // ---- S = Q*K^T : batch B-operand loads, then WMMAs (overlap LDS latency) ----
    v16bf kb[4];
    v8f st[4];
#pragma unroll
    for (int t = 0; t < 4; ++t)
      kb[t] = ldsB(&kv_row[(16*t + l15) * KV_STRIDE + doff]);
#pragma unroll
    for (int t = 0; t < 4; ++t) {
      v8f z = {};
      st[t] = WMMA_BF16(qa[0].v, kb[t], z);
    }
#pragma unroll
    for (int t = 0; t < 4; ++t)
      kb[t] = ldsB(&kv_row[(16*t + l15) * KV_STRIDE + 32 + doff]);
#pragma unroll
    for (int t = 0; t < 4; ++t)
      st[t] = WMMA_BF16(qa[1].v, kb[t], st[t]);

    // ---- online softmax (multiplicative mask BEFORE softmax) ----
    __bf16* pp = &p_lds[(wid * QW) * P_STRIDE];
    float fac[8];
#pragma unroll
    for (int r = 0; r < 8; ++r) {
      float a0 = st[0][r] * mf[r], a1 = st[1][r] * mf[r];
      float a2 = st[2][r] * mf[r], a3 = st[3][r] * mf[r];
      float mx = hmax16(fmaxf(fmaxf(a0, a1), fmaxf(a2, a3)));
      float mnew = fmaxf(mrow[r], mx);
      fac[r] = fexp2(mrow[r] - mnew);
      mrow[r] = mnew;
      float p0 = fexp2(a0 - mnew), p1 = fexp2(a1 - mnew);
      float p2 = fexp2(a2 - mnew), p3 = fexp2(a3 - mnew);
      const int prow = r + 8 * hi;          // C layout -> (row, key) in LDS
      pp[prow * P_STRIDE + l15]      = (__bf16)p0;
      pp[prow * P_STRIDE + 16 + l15] = (__bf16)p1;
      pp[prow * P_STRIDE + 32 + l15] = (__bf16)p2;
      pp[prow * P_STRIDE + 48 + l15] = (__bf16)p3;
    }
#pragma unroll
    for (int r = 0; r < 8; ++r) {           // rescale O and row-sum accumulator
      float f = fac[r];
      o0[r] *= f; o1[r] *= f; o2[r] *= f; o3[r] *= f; ls[r] *= f;
    }

    // ---- reload P (16 x 64 keys) in WMMA A layout: two k=32 slices ----
    V16U pa0, pa1;
    {
      const __bf16* pr = &p_lds[(wid * QW + l15) * P_STRIDE + aoff];
      pa0.u[0] = *(const uint4*)pr;
      pa0.u[1] = *(const uint4*)(pr + 16);
      pa1.u[0] = *(const uint4*)(pr + 32);
      pa1.u[1] = *(const uint4*)(pr + 48);
    }

    // ---- O += P*V : 4 dim groups x 2 key slices (batched B loads) ----
    v16bf vb[4];
#pragma unroll
    for (int g = 0; g < 4; ++g)
      vb[g] = ldsB(&kv_tr[(16*g + l15) * TR_STRIDE + doff]);
    o0 = WMMA_BF16(pa0.v, vb[0], o0);
    o1 = WMMA_BF16(pa0.v, vb[1], o1);
    o2 = WMMA_BF16(pa0.v, vb[2], o2);
    o3 = WMMA_BF16(pa0.v, vb[3], o3);
#pragma unroll
    for (int g = 0; g < 4; ++g)
      vb[g] = ldsB(&kv_tr[(16*g + l15) * TR_STRIDE + 32 + doff]);
    o0 = WMMA_BF16(pa1.v, vb[0], o0);
    o1 = WMMA_BF16(pa1.v, vb[1], o1);
    o2 = WMMA_BF16(pa1.v, vb[2], o2);
    o3 = WMMA_BF16(pa1.v, vb[3], o3);

    // ---- row sums on the matrix pipe: ls += P * ones ----
    ls = WMMA_BF16(pa0.v, onesb.v, ls);
    ls = WMMA_BF16(pa1.v, onesb.v, ls);
  }

  // ---- normalize and store (C layout: row = r + 8*hi, col = 16g + l15) ----
#pragma unroll
  for (int r = 0; r < 8; ++r) {
    float inv = 1.0f / ls[r];
    size_t base = (size_t)(b * T_SEQ + q0 + r + 8 * hi) * DMODEL + h * HD + l15;
    out[base]      = o0[r] * inv;
    out[base + 16] = o1[r] * inv;
    out[base + 32] = o2[r] * inv;
    out[base + 48] = o3[r] * inv;
  }
}

extern "C" void kernel_launch(void* const* d_in, const int* in_sizes, int n_in,
                              void* d_out, int out_size, void* d_ws, size_t ws_size,
                              hipStream_t stream) {
  const float* x = (const float*)d_in[0];
  const float* w = (const float*)d_in[1];
  float* out = (float*)d_out;
  const int B = in_sizes[0] / (T_SEQ * DMODEL);   // 2
  dim3 grid(B * NH * (T_SEQ / QB));               // 1024 blocks
  dim3 block(128);                                // 4 wave32
  hipLaunchKernelGGL(palace_attn_wmma, grid, block, 0, stream, x, w, out);
}
